// Block_58789512348490
// MI455X (gfx1250) — matrix operand order repaired and checked by reading, not compiled, not run on hardware
//
#include <hip/hip_runtime.h>

typedef unsigned short u16;
typedef __attribute__((ext_vector_type(16))) __bf16 v16bf;
typedef __attribute__((ext_vector_type(8)))  float  v8f;

#define NPTS 16384
#define CDIM 256
#define KNB  16
#define DIN  512
#define DST  16
#define DTRK 16
#define DCV  4

enum { F_LN = 1, F_RELU = 2, F_BIAS = 4, F_RPRE = 8, F_RPOST = 16 };

__device__ __forceinline__ u16 f2bf(float f) {
  unsigned u = __float_as_uint(f);
  u += 0x7FFFu + ((u >> 16) & 1u);      // round-to-nearest-even
  return (u16)(u >> 16);
}
__device__ __forceinline__ float siluf_(float x) { return x / (1.f + __expf(-x)); }
__device__ __forceinline__ float softplusf_(float x) {
  return (x > 20.f) ? x : log1pf(__expf(x));
}

// ---------------------------------------------------------------- convert
__global__ void cvt_f32_bf16(const float* __restrict__ in, u16* __restrict__ out, int n) {
  int i = blockIdx.x * blockDim.x + threadIdx.x;
  if (i < n) out[i] = f2bf(in[i]);
}

// ---------------------------------------------------------------- WMMA GEMM
// out[M,NC] = epilogue( A[M,K](bf16) @ W[NC,K]^T(bf16) )
// block = 256 threads = 8 waves; block computes 16 rows x 256 cols.
union Frag { uint4 q[2]; v16bf v; };

__global__ __launch_bounds__(256) void wmma_gemm(
    const u16* __restrict__ A, const u16* __restrict__ W, int K, int NCOLS,
    const float* __restrict__ bias, const float* __restrict__ rpre,
    const float* __restrict__ rpost,
    const float* __restrict__ lng, const float* __restrict__ lnb,
    float* __restrict__ outF, int ldF,
    u16* __restrict__ outB, int ldB, int flags)
{
  const int lane = threadIdx.x & 31;
  const int wave = threadIdx.x >> 5;
  const int half = lane >> 4;          // K half selector
  const int l16  = lane & 15;
  const int rowBase = blockIdx.x << 4;
  const int colBase = (blockIdx.y << 8) + (wave << 5);

  v8f acc0 = {};
  v8f acc1 = {};

  if (colBase < NCOLS) {
    const u16* aRow = A + (size_t)(rowBase + l16) * K + half * 16;
    int c0 = colBase + l16;       if (c0 > NCOLS - 1) c0 = NCOLS - 1;
    int c1 = colBase + 16 + l16;  if (c1 > NCOLS - 1) c1 = NCOLS - 1;
    const u16* w0 = W + (size_t)c0 * K + half * 16;
    const u16* w1 = W + (size_t)c1 * K + half * 16;
    for (int k = 0; k < K; k += 32) {
      Frag fa, fb0, fb1;
      fa.q[0]  = *(const uint4*)(aRow + k);        // K = half*16 .. +7
      fa.q[1]  = *(const uint4*)(aRow + k + 8);    // K = half*16+8 .. +15
      fb0.q[0] = *(const uint4*)(w0 + k);          // K = half*16 .. +7
      fb0.q[1] = *(const uint4*)(w0 + k + 8);      // K = half*16+8 .. +15
      fb1.q[0] = *(const uint4*)(w1 + k);
      fb1.q[1] = *(const uint4*)(w1 + k + 8);
      acc0 = __builtin_amdgcn_wmma_f32_16x16x32_bf16(false, fa.v, false, fb0.v,
                                                     (short)0, acc0, false, false);
      acc1 = __builtin_amdgcn_wmma_f32_16x16x32_bf16(false, fa.v, false, fb1.v,
                                                     (short)0, acc1, false, false);
    }
  }

  __shared__ float tile[16][264];
  __shared__ float ps[16][17], ps2[16][17];
  __shared__ float mu[16], rs[16];

  if (flags & F_LN) {              // requires NCOLS==256, gridDim.y==1
    #pragma unroll
    for (int v = 0; v < 8; ++v) {
      int r = v + (half << 3);
      int grow = rowBase + r;
      int cc0 = colBase + l16, cc1 = colBase + 16 + l16;
      float x0 = acc0[v], x1 = acc1[v];
      if (flags & F_BIAS) { x0 += bias[cc0]; x1 += bias[cc1]; }
      if (flags & F_RPRE) {
        x0 += rpre[(size_t)grow * CDIM + cc0];
        x1 += rpre[(size_t)grow * CDIM + cc1];
      }
      tile[r][cc0] = x0; tile[r][cc1] = x1;
    }
    __syncthreads();
    {
      int r = threadIdx.x >> 4, seg = threadIdx.x & 15;
      float s = 0.f, s2 = 0.f;
      #pragma unroll
      for (int i = 0; i < 16; ++i) { float x = tile[r][(seg << 4) + i]; s += x; s2 += x * x; }
      ps[r][seg] = s; ps2[r][seg] = s2;
    }
    __syncthreads();
    if (threadIdx.x < 16) {
      int r = threadIdx.x; float s = 0.f, s2 = 0.f;
      #pragma unroll
      for (int i = 0; i < 16; ++i) { s += ps[r][i]; s2 += ps2[r][i]; }
      float m = s * (1.f / 256.f);
      float var = s2 * (1.f / 256.f) - m * m;
      mu[r] = m; rs[r] = rsqrtf(var + 1e-5f);
    }
    __syncthreads();
    {
      int r = threadIdx.x >> 4, seg = threadIdx.x & 15;
      int grow = rowBase + r;
      #pragma unroll
      for (int i = 0; i < 16; ++i) {
        int col = (seg << 4) + i;
        float y = (tile[r][col] - mu[r]) * rs[r] * lng[col] + lnb[col];
        if (flags & F_RELU)  y = fmaxf(y, 0.f);
        if (flags & F_RPOST) y = fmaxf(y + rpost[(size_t)grow * CDIM + col], 0.f);
        if (outF) outF[(size_t)grow * ldF + col] = y;
        if (outB) outB[(size_t)grow * ldB + col] = f2bf(y);
      }
    }
  } else if (colBase < NCOLS) {
    #pragma unroll
    for (int v = 0; v < 8; ++v) {
      int grow = rowBase + v + (half << 3);
      int cc0 = colBase + l16, cc1 = colBase + 16 + l16;
      float x0 = acc0[v], x1 = acc1[v];
      if ((flags & F_BIAS) && cc0 < NCOLS) x0 += bias[cc0];
      if ((flags & F_BIAS) && cc1 < NCOLS) x1 += bias[cc1];
      if (outF) {
        if (cc0 < NCOLS) outF[(size_t)grow * ldF + cc0] = x0;
        if (cc1 < NCOLS) outF[(size_t)grow * ldF + cc1] = x1;
      }
      if (outB) {
        if (cc0 < NCOLS) outB[(size_t)grow * ldB + cc0] = f2bf(x0);
        if (cc1 < NCOLS) outB[(size_t)grow * ldB + cc1] = f2bf(x1);
      }
    }
  }
}

// ---------------------------------------------------------------- gather + LN
__global__ __launch_bounds__(256) void gather_ln(
    const float* __restrict__ f1, const float* __restrict__ gd, const int* __restrict__ idx,
    const float* __restrict__ lng, const float* __restrict__ lnb,
    u16* __restrict__ out, int ld)
{
  int n = blockIdx.x, c = threadIdx.x;
  float acc = 0.f;
  #pragma unroll
  for (int k = 0; k < KNB; ++k) {
    int j = idx[n * KNB + k];
    acc += gd[n * KNB + k] * f1[(size_t)j * CDIM + c];
  }
  __shared__ float r1[256], r2[256];
  r1[c] = acc; r2[c] = acc * acc;
  __syncthreads();
  for (int s = 128; s > 0; s >>= 1) {
    if (c < s) { r1[c] += r1[c + s]; r2[c] += r2[c + s]; }
    __syncthreads();
  }
  float m = r1[0] * (1.f / 256.f);
  float var = r2[0] * (1.f / 256.f) - m * m;
  float y = (acc - m) * rsqrtf(var + 1e-5f) * lng[c] + lnb[c];
  out[(size_t)n * ld + c] = f2bf(y);
}

// ---------------------------------------------------------------- rmsnorm
__global__ __launch_bounds__(256) void rmsnorm_k(
    const float* __restrict__ x, const float* __restrict__ w, u16* __restrict__ out)
{
  int n = blockIdx.x, c = threadIdx.x;
  float v = x[(size_t)n * CDIM + c];
  __shared__ float r[256];
  r[c] = v * v;
  __syncthreads();
  for (int s = 128; s > 0; s >>= 1) { if (c < s) r[c] += r[c + s]; __syncthreads(); }
  float ms = r[0] * (1.f / 256.f);
  out[(size_t)n * CDIM + c] = f2bf(v * rsqrtf(ms + 1e-5f) * w[c]);
}

// ---------------------------------------------------------------- depthwise conv4 + silu
__global__ void conv_silu(const float* __restrict__ xz, const float* __restrict__ w,
                          const float* __restrict__ bias,
                          float* __restrict__ xmF, u16* __restrict__ xmB)
{
  int i = blockIdx.x * blockDim.x + threadIdx.x;
  if (i >= NPTS * DIN) return;
  int n = i >> 9, d = i & (DIN - 1);
  float s = bias[d];
  #pragma unroll
  for (int j = 0; j < DCV; ++j) {
    int m = n - (DCV - 1) + j;
    if (m >= 0) s += xz[(size_t)m * (2 * DIN) + d] * w[d * DCV + j];
  }
  float y = siluf_(s);
  xmF[i] = y; xmB[i] = f2bf(y);
}

// ---------------------------------------------------------------- dt projection + softplus
__global__ void dt_softplus(const float* __restrict__ dbc, const float* __restrict__ w,
                            const float* __restrict__ b, float* __restrict__ delta)
{
  int i = blockIdx.x * blockDim.x + threadIdx.x;
  if (i >= NPTS * DIN) return;
  int n = i >> 9, d = i & (DIN - 1);
  float s = b[d];
  #pragma unroll
  for (int t = 0; t < DTRK; ++t) s += dbc[n * 48 + t] * w[d * DTRK + t];
  delta[i] = softplusf_(s);
}

// ---------------------------------------------------------------- selective scan
// one wave32 per channel d; lanes 0..15 hold the DS=16 state
__global__ __launch_bounds__(256) void sel_scan(
    const float* __restrict__ delta, const float* __restrict__ dbc,
    const float* __restrict__ u, const float* __restrict__ A_log,
    float* __restrict__ ys)
{
  int lane = threadIdx.x & 31;
  int wave = threadIdx.x >> 5;
  int d = (blockIdx.x << 3) + wave;
  int s = lane & 15;
  float A = -__expf(A_log[d * DST + s]);
  float h = 0.f;
  for (int n = 0; n < NPTS; ++n) {
    float dt = delta[(size_t)n * DIN + d];
    float uu = u[(size_t)n * DIN + d];
    float Bn = dbc[n * 48 + DTRK + s];
    float Cn = dbc[n * 48 + DTRK + DST + s];
    h = __expf(dt * A) * h + dt * uu * Bn;
    float p = h * Cn;
    p += __shfl_xor(p, 8, 32);
    p += __shfl_xor(p, 4, 32);
    p += __shfl_xor(p, 2, 32);
    p += __shfl_xor(p, 1, 32);
    if (lane == 0) ys[(size_t)n * DIN + d] = p;
    if ((n & 7) == 0 && n + 16 < NPTS) {
      __builtin_prefetch(&delta[(size_t)(n + 16) * DIN + d], 0, 0);
      __builtin_prefetch(&u[(size_t)(n + 16) * DIN + d], 0, 0);
    }
  }
}

// ---------------------------------------------------------------- gate: (ys + xm*D) * silu(z)
__global__ void y_gate(const float* __restrict__ ys, const float* __restrict__ xm,
                       const float* __restrict__ Dp, const float* __restrict__ xz,
                       u16* __restrict__ yB)
{
  int i = blockIdx.x * blockDim.x + threadIdx.x;
  if (i >= NPTS * DIN) return;
  int n = i >> 9, d = i & (DIN - 1);
  float z = xz[(size_t)n * (2 * DIN) + DIN + d];
  float y = (ys[i] + xm[i] * Dp[d]) * siluf_(z);
  yB[i] = f2bf(y);
}

// ================================================================ host
extern "C" void kernel_launch(void* const* d_in, const int* in_sizes, int n_in,
                              void* d_out, int out_size, void* d_ws, size_t ws_size,
                              hipStream_t stream)
{
  (void)in_sizes; (void)n_in; (void)out_size; (void)ws_size;
  const float* feat     = (const float*)d_in[0];
  const float* gauss    = (const float*)d_in[2];
  const int*   ridx     = (const int*)  d_in[3];
  const float* fc1_w    = (const float*)d_in[4];
  const float* fc3_w    = (const float*)d_in[5];
  const float* ln1_g    = (const float*)d_in[6];
  const float* ln1_b    = (const float*)d_in[7];
  const float* ln2_g    = (const float*)d_in[8];
  const float* ln2_b    = (const float*)d_in[9];
  const float* ln3_g    = (const float*)d_in[10];
  const float* ln3_b    = (const float*)d_in[11];
  const float* attn_g   = (const float*)d_in[12];
  const float* attn_b   = (const float*)d_in[13];
  const float* la_w1    = (const float*)d_in[14];
  const float* la_b1    = (const float*)d_in[15];
  const float* la_ln_g  = (const float*)d_in[16];
  const float* la_ln_b  = (const float*)d_in[17];
  const float* la_w2    = (const float*)d_in[18];
  const float* la_b2    = (const float*)d_in[19];
  const float* rms_w    = (const float*)d_in[20];
  const float* inproj_w = (const float*)d_in[21];
  const float* conv_w   = (const float*)d_in[22];
  const float* conv_b   = (const float*)d_in[23];
  const float* xproj_w  = (const float*)d_in[24];
  const float* dtproj_w = (const float*)d_in[25];
  const float* dtproj_b = (const float*)d_in[26];
  const float* A_log    = (const float*)d_in[27];
  const float* D_param  = (const float*)d_in[28];
  const float* outproj_w= (const float*)d_in[29];

  char* p = (char*)d_ws;
  auto alloc = [&](size_t bytes) -> void* {
    void* r = (void*)p; p += (bytes + 255) & ~(size_t)255; return r;
  };

  u16* feat_bf = (u16*)alloc((size_t)NPTS * CDIM * 2);
  u16* fc1_bf  = (u16*)alloc((size_t)CDIM * CDIM * 2);
  u16* fc3_bf  = (u16*)alloc((size_t)CDIM * CDIM * 2);
  u16* la1_bf  = (u16*)alloc((size_t)CDIM * 2 * CDIM * 2);
  u16* la2_bf  = (u16*)alloc((size_t)CDIM * CDIM * 2);
  u16* inp_bf  = (u16*)alloc((size_t)2 * DIN * CDIM * 2);
  u16* xp_bf   = (u16*)alloc((size_t)48 * DIN * 2);
  u16* op_bf   = (u16*)alloc((size_t)CDIM * DIN * 2);
  float* f1    = (float*)alloc((size_t)NPTS * CDIM * 4);
  u16*  xr_bf  = (u16*) alloc((size_t)NPTS * CDIM * 2);
  float* xz    = (float*)alloc((size_t)NPTS * 2 * DIN * 4);
  float* xmF   = (float*)alloc((size_t)NPTS * DIN * 4);
  u16*  xm_bf  = (u16*) alloc((size_t)NPTS * DIN * 2);
  float* dbc   = (float*)alloc((size_t)NPTS * 48 * 4);
  float* delta = (float*)alloc((size_t)NPTS * DIN * 4);
  float* ys    = (float*)alloc((size_t)NPTS * DIN * 4);
  u16*  y_bf   = (u16*) alloc((size_t)NPTS * DIN * 2);
  u16*  cat_bf = (u16*) alloc((size_t)NPTS * 2 * CDIM * 2);
  u16*  res_bf = (u16*) alloc((size_t)NPTS * CDIM * 2);
  u16*  f2_bf  = (u16*) alloc((size_t)NPTS * CDIM * 2);

  auto cvt = [&](const float* src, u16* dst, int n) {
    cvt_f32_bf16<<<(n + 255) / 256, 256, 0, stream>>>(src, dst, n);
  };
  auto gemm = [&](const u16* A, const u16* W, int K, int NC,
                  const float* bias, const float* rpre, const float* rpost,
                  const float* lg, const float* lb,
                  float* oF, int ldF, u16* oB, int ldB, int flags) {
    dim3 grid(NPTS / 16, (NC + 255) / 256);
    wmma_gemm<<<grid, 256, 0, stream>>>(A, W, K, NC, bias, rpre, rpost,
                                        lg, lb, oF, ldF, oB, ldB, flags);
  };

  // convert activations + weights to bf16 once
  cvt(feat,      feat_bf, NPTS * CDIM);
  cvt(fc1_w,     fc1_bf,  CDIM * CDIM);
  cvt(fc3_w,     fc3_bf,  CDIM * CDIM);
  cvt(la_w1,     la1_bf,  CDIM * 2 * CDIM);
  cvt(la_w2,     la2_bf,  CDIM * CDIM);
  cvt(inproj_w,  inp_bf,  2 * DIN * CDIM);
  cvt(xproj_w,   xp_bf,   48 * DIN);
  cvt(outproj_w, op_bf,   CDIM * DIN);

  // f1 = relu(LN1(feat @ fc1^T))
  gemm(feat_bf, fc1_bf, CDIM, CDIM, nullptr, nullptr, nullptr,
       ln1_g, ln1_b, f1, CDIM, nullptr, 0, F_LN | F_RELU);

  // combined = LN_attn(einsum) -> cat[:,256:512]
  gather_ln<<<NPTS, 256, 0, stream>>>(f1, gauss, ridx, attn_g, attn_b,
                                      cat_bf + CDIM, 2 * CDIM);
  // xr = rmsnorm(f1)
  rmsnorm_k<<<NPTS, 256, 0, stream>>>(f1, rms_w, xr_bf);

  // xz = xr @ in_proj^T  (N x 1024)
  gemm(xr_bf, inp_bf, CDIM, 2 * DIN, nullptr, nullptr, nullptr,
       nullptr, nullptr, xz, 2 * DIN, nullptr, 0, 0);

  // xm = silu(causal depthwise conv4(xz[:, :512]))
  conv_silu<<<(NPTS * DIN + 255) / 256, 256, 0, stream>>>(xz, conv_w, conv_b, xmF, xm_bf);

  // dbc = xm @ x_proj^T  (N x 48)
  gemm(xm_bf, xp_bf, DIN, 48, nullptr, nullptr, nullptr,
       nullptr, nullptr, dbc, 48, nullptr, 0, 0);

  // delta = softplus(dbc[:, :16] @ dt_proj^T + b)
  dt_softplus<<<(NPTS * DIN + 255) / 256, 256, 0, stream>>>(dbc, dtproj_w, dtproj_b, delta);

  // selective scan (one wave per channel)
  sel_scan<<<DIN / 8, 256, 0, stream>>>(delta, dbc, xmF, A_log, ys);

  // y = (ys + xm*D) * silu(z)
  y_gate<<<(NPTS * DIN + 255) / 256, 256, 0, stream>>>(ys, xmF, D_param, xz, y_bf);

  // feat_mamba = LN_attn(y @ out_proj^T + f1) -> cat[:, :256]
  gemm(y_bf, op_bf, DIN, CDIM, nullptr, f1, nullptr,
       attn_g, attn_b, nullptr, 0, cat_bf, 2 * CDIM, F_LN | F_RPRE);

  // res = relu(LN_la(cat @ la_w1^T + la_b1))
  gemm(cat_bf, la1_bf, 2 * CDIM, CDIM, la_b1, nullptr, nullptr,
       la_ln_g, la_ln_b, nullptr, 0, res_bf, CDIM, F_LN | F_RELU | F_BIAS);

  // f2 = relu(LN2(res @ la_w2^T + la_b2))
  gemm(res_bf, la2_bf, CDIM, CDIM, la_b2, nullptr, nullptr,
       ln2_g, ln2_b, nullptr, 0, f2_bf, CDIM, F_LN | F_RELU | F_BIAS);

  // out = relu(feat + LN3(f2 @ fc3^T))
  gemm(f2_bf, fc3_bf, CDIM, CDIM, nullptr, nullptr, feat,
       ln3_g, ln3_b, (float*)d_out, CDIM, nullptr, 0, F_LN | F_RPOST);
}
